// MS_Attention_linear_3d_23313082483195
// MI455X (gfx1250) — matrix operand order repaired and checked by reading, not compiled, not run on hardware
//
#include <hip/hip_runtime.h>
#include <hip/hip_bf16.h>

// ---------------------------------------------------------------------------
// MS linear spiking attention for MI455X (gfx1250, wave32, WMMA).
// All GEMMs: V_WMMA_F32_16X16X32_F16. Spikes {0,1} exact in f16; kv entries
// are integers <= 576 (exact in f16). Global->LDS via async-to-LDS (ASYNCcnt)
// with double buffering; two barriers per 32-wide K step; 4 WMMA/wave/step.
// Layout trick: x/q/o spikes stored pixel-major [n][c], k/v spikes stored
// channel-major [c][n] so every GEMM tile load is K-contiguous (no transpose).
// ---------------------------------------------------------------------------

#ifndef USE_ASYNC
#define USE_ASYNC 1
#endif

typedef __attribute__((ext_vector_type(16))) _Float16 v16h;
typedef __attribute__((ext_vector_type(8)))  _Float16 v8h;
typedef __attribute__((ext_vector_type(8)))  float    v8f;

#define T_DIM 4
#define B_DIM 8
#define C_DIM 512
#define N_PIX 576
#define CV_DIM 2048
#define HEADS 8
#define DH 64
#define DHV 256

// LDS byte-offset of a __shared__ object (generic -> addrspace(3) cast).
__device__ __forceinline__ unsigned lds_off(const void* p) {
    return (unsigned)(unsigned long long)
        ((__attribute__((address_space(3))) const void*)p);
}

#if USE_ASYNC
#define ASYNC_B128(lds, gaddr)                                              \
    asm volatile("global_load_async_to_lds_b128 %0, %1, off"                \
                 ::"v"(lds), "v"(gaddr) : "memory")
#define ASYNC_B128_O16(lds, gaddr)                                          \
    asm volatile("global_load_async_to_lds_b128 %0, %1, off offset:16"      \
                 ::"v"(lds), "v"(gaddr) : "memory")
#define WAIT_ASYNC() asm volatile("s_wait_asynccnt 0x0" ::: "memory")
#endif

// --------------------------- f32 -> f16 convert ----------------------------
__global__ __launch_bounds__(256) void cvt_f16_kernel(const float* __restrict__ in,
                                                      _Float16* __restrict__ out,
                                                      long n) {
    long i = (long)blockIdx.x * blockDim.x + threadIdx.x;
    if (i < n) out[i] = (_Float16)in[i];
}

// ------------------------------- LIF neuron --------------------------------
// in: [T, B, CH, N] f32 (+ optional pos[CH,N]); out f16 spikes either in the
// same layout (TRANS=false) or pixel-major [T, B, N, CH] (TRANS=true).
template <bool TRANS>
__global__ __launch_bounds__(256) void lif_kernel(const float* __restrict__ in,
                                                  const float* __restrict__ pos,
                                                  _Float16* __restrict__ out,
                                                  long per_t, int CH) {
    long idx = (long)blockIdx.x * blockDim.x + threadIdx.x;
    if (idx >= per_t) return;
    const long chn = (long)CH * N_PIX;
    const long cn = idx % chn;                 // c*N + n
    float p = pos ? pos[cn] : 0.0f;
    long obase = idx;
    if (TRANS) {
        long b = idx / chn;
        int c = (int)(cn / N_PIX);
        int n = (int)(cn % N_PIX);
        obase = (b * N_PIX + n) * CH + c;
    }
    float mem = 0.0f, s = 0.0f;
    #pragma unroll
    for (int t = 0; t < T_DIM; ++t) {
        float xt = in[(long)t * per_t + idx] + p;
        mem = mem * 0.25f * (1.0f - s) + xt;
        s = (mem > 0.5f) ? 1.0f : 0.0f;
        out[(long)t * per_t + obase] = (_Float16)s;
    }
}

// ------------------------------ WMMA GEMM ----------------------------------
// C[m,n] = alpha * sum_k A[m,k]*B[k,n]   (+ optional per-row s[m], b[m])
//   A: f16 [M x K] row-major (lda), batch offset z*strideA
//   B: f16 stored "N-major": Bg[n*ldb + k]; batch offset
//      (z/headsB)*strideB_o + (z%headsB)*strideB_i
// Block: 256 thr = 8 waves; tile M=128 x N=64; wave -> 32x32 (2x2 WMMA);
// K-step 32; double-buffered LDS, two barriers + one asynccnt wait per step.
template <bool OUT_F16, bool HAS_SB>
__global__ __launch_bounds__(256) void gemm_wmma_kernel(
    const _Float16* __restrict__ A, long strideA,
    const _Float16* __restrict__ Bg, long strideB_o, long strideB_i, int headsB,
    void* __restrict__ Cg, long strideC,
    int Kdim, int lda, int ldb, int ldc,
    const float* __restrict__ srow, const float* __restrict__ brow,
    float alpha) {
    // Rows padded to 48 halves (96B) -> all fragment reads are 16B-aligned
    // ds_load_b128 and bank patterns are spread.
    __shared__ __align__(16) _Float16 lA[2][128][48];
    __shared__ __align__(16) _Float16 lB[2][64][48];

    const int tid  = threadIdx.x;
    const int lane = tid & 31;
    const int wave = tid >> 5;
    const int wm   = (wave & 3) * 32;
    const int wn   = (wave >> 2) * 32;

    const int m0 = blockIdx.x * 128;
    const int n0 = blockIdx.y * 64;
    const int z  = blockIdx.z;
    const int tb = z / headsB;
    const int hh = z - tb * headsB;

    const _Float16* Ab = A  + (long)z * strideA;
    const _Float16* Bb = Bg + (long)tb * strideB_o + (long)hh * strideB_i;

    // Cooperative-load thread mapping (K-contiguous 16B chunks).
    const int ra = tid >> 1;            // 0..127  (A row = m)
    const int ca = (tid & 1) * 16;      // 0 / 16  (A col = k), +8 via offset:16
    const int rb = tid >> 2;            // 0..63   (B row = n)
    const int cb = (tid & 3) * 8;       // 0,8,16,24 (B col = k)

    unsigned long long gA =
        (unsigned long long)(const void*)(Ab + (long)(m0 + ra) * lda + ca);
    unsigned long long gB =
        (unsigned long long)(const void*)(Bb + (long)(n0 + rb) * ldb + cb);

    // Fragment lane mapping (ISA 7.12.2, 16-bit A 16x32 / B 32x16):
    const int mA   = lane & 15;
    const int kloA = (lane < 16) ? 0 : 8;
    const int nB   = lane & 15;
    const int kloB = (lane < 16) ? 0 : 8;

    v8f acc[2][2];
    #pragma unroll
    for (int i = 0; i < 2; ++i)
        #pragma unroll
        for (int j = 0; j < 2; ++j)
            acc[i][j] = (v8f){0.f, 0.f, 0.f, 0.f, 0.f, 0.f, 0.f, 0.f};

    const int nk = Kdim >> 5;

#if USE_ASYNC
    const unsigned ldsA0 = lds_off(&lA[0][ra][ca]);
    const unsigned ldsA1 = lds_off(&lA[1][ra][ca]);
    const unsigned ldsB0 = lds_off(&lB[0][rb][cb]);
    const unsigned ldsB1 = lds_off(&lB[1][rb][cb]);
    // Prologue: async tile 0 -> buffer 0.
    ASYNC_B128(ldsA0, gA);
    ASYNC_B128_O16(ldsA0, gA);
    ASYNC_B128(ldsB0, gB);
#else
    v8h sa0, sa1, sb0;
    sa0 = *(const v8h*)(const void*)gA;
    sa1 = *(const v8h*)(const void*)(gA + 16);
    sb0 = *(const v8h*)(const void*)gB;
#endif

    for (int i = 0; i < nk; ++i) {
        const int cur = i & 1;
#if USE_ASYNC
        WAIT_ASYNC();             // my async stores for tile i landed in LDS
        __syncthreads();          // => everyone's tile i complete & visible
        if (i + 1 < nk) {
            unsigned long long a1 = gA + (unsigned long long)(i + 1) * 64u;
            unsigned long long b1 = gB + (unsigned long long)(i + 1) * 64u;
            const unsigned la = (i & 1) ? ldsA0 : ldsA1;
            const unsigned lb = (i & 1) ? ldsB0 : ldsB1;
            ASYNC_B128(la, a1);
            ASYNC_B128_O16(la, a1);
            ASYNC_B128(lb, b1);
        }
#else
        *(v8h*)&lA[cur][ra][ca]     = sa0;
        *(v8h*)&lA[cur][ra][ca + 8] = sa1;
        *(v8h*)&lB[cur][rb][cb]     = sb0;
        __syncthreads();
        if (i + 1 < nk) {
            unsigned long long a1 = gA + (unsigned long long)(i + 1) * 64u;
            unsigned long long b1 = gB + (unsigned long long)(i + 1) * 64u;
            sa0 = *(const v8h*)(const void*)a1;
            sa1 = *(const v8h*)(const void*)(a1 + 16);
            sb0 = *(const v8h*)(const void*)b1;
        }
#endif
        // ---- fragments ----
        v16h af[2], bf[2];
        #pragma unroll
        for (int s = 0; s < 2; ++s) {
            const _Float16* ap = &lA[cur][wm + s * 16 + mA][0];
            v8h alo = *(const v8h*)(ap + kloA);
            v8h ahi = *(const v8h*)(ap + kloA + 16);
            af[s] = __builtin_shufflevector(alo, ahi, 0, 1, 2, 3, 4, 5, 6, 7,
                                            8, 9, 10, 11, 12, 13, 14, 15);
            const _Float16* bp = &lB[cur][wn + s * 16 + nB][0];
            v8h blo = *(const v8h*)(bp + kloB);
            v8h bhi = *(const v8h*)(bp + kloB + 16);
            bf[s] = __builtin_shufflevector(blo, bhi, 0, 1, 2, 3, 4, 5, 6, 7,
                                            8, 9, 10, 11, 12, 13, 14, 15);
        }
        // ---- 2x2 WMMA ----
        #pragma unroll
        for (int sm = 0; sm < 2; ++sm)
            #pragma unroll
            for (int sn = 0; sn < 2; ++sn)
                acc[sm][sn] = __builtin_amdgcn_wmma_f32_16x16x32_f16(
                    false, af[sm], false, bf[sn], (short)0, acc[sm][sn],
                    false, false);
        __syncthreads();
    }

    // ---- epilogue: C/D layout: VGPR r -> M = r + 8*(lane>=16), N = lane&15
    #pragma unroll
    for (int sm = 0; sm < 2; ++sm) {
        const int mbase = m0 + wm + sm * 16 + ((lane >> 4) << 3);
        #pragma unroll
        for (int sn = 0; sn < 2; ++sn) {
            const int ncol = n0 + wn + sn * 16 + (lane & 15);
            #pragma unroll
            for (int r = 0; r < 8; ++r) {
                const int m = mbase + r;
                float val = acc[sm][sn][r] * alpha;
                if (HAS_SB) val = val * srow[m] + brow[m];
                if (OUT_F16) {
                    ((_Float16*)Cg)[(long)z * strideC + (long)m * ldc + ncol] =
                        (_Float16)val;
                } else {
                    ((float*)Cg)[(long)z * strideC + (long)m * ldc + ncol] = val;
                }
            }
        }
    }
}

// ---------------------------------------------------------------------------
extern "C" void kernel_launch(void* const* d_in, const int* in_sizes, int n_in,
                              void* d_out, int out_size, void* d_ws,
                              size_t ws_size, hipStream_t stream) {
    (void)in_sizes; (void)n_in; (void)out_size; (void)ws_size;

    const float* x_in = (const float*)d_in[0];
    const float* pos  = (const float*)d_in[1];
    const float* q_w  = (const float*)d_in[2];
    const float* q_s  = (const float*)d_in[3];
    const float* q_b  = (const float*)d_in[4];
    const float* k_w  = (const float*)d_in[5];
    const float* k_s  = (const float*)d_in[6];
    const float* k_b  = (const float*)d_in[7];
    const float* v_w  = (const float*)d_in[8];
    const float* v_s  = (const float*)d_in[9];
    const float* v_b  = (const float*)d_in[10];
    const float* p_w  = (const float*)d_in[11];
    const float* p_s  = (const float*)d_in[12];
    const float* p_b  = (const float*)d_in[13];
    float* out = (float*)d_out;

    const long TBCN  = (long)T_DIM * B_DIM * C_DIM * N_PIX;
    const long TBCvN = (long)T_DIM * B_DIM * CV_DIM * N_PIX;
    const long CN    = (long)C_DIM * N_PIX;
    const long CvN   = (long)CV_DIM * N_PIX;
    const int  TB    = T_DIM * B_DIM;     // 32
    const int  TBH   = TB * HEADS;        // 256

    // ----------------------------- workspace ------------------------------
    char* ws = (char*)d_ws;
    size_t off = 0;
    auto take = [&](size_t bytes) {
        size_t r = off;
        off += (bytes + 255) & ~(size_t)255;
        return r;
    };
    _Float16* wq16 = (_Float16*)(ws + take((size_t)C_DIM * C_DIM * 2));
    _Float16* wk16 = (_Float16*)(ws + take((size_t)C_DIM * C_DIM * 2));
    _Float16* wv16 = (_Float16*)(ws + take((size_t)CV_DIM * C_DIM * 2));
    _Float16* wp16 = (_Float16*)(ws + take((size_t)C_DIM * CV_DIM * 2));
    _Float16* xsT  = (_Float16*)(ws + take((size_t)TBCN * 2));   // [tb][n][c]
    _Float16* qsT  = (_Float16*)(ws + take((size_t)TBCN * 2));   // [tb][n][c]
    _Float16* ks   = (_Float16*)(ws + take((size_t)TBCN * 2));   // [tb][c][n]
    _Float16* vs   = (_Float16*)(ws + take((size_t)TBCvN * 2));  // [tb][c][n]
    _Float16* vkb  = (_Float16*)(ws + take((size_t)TBH * DHV * DH * 2));
    // Big f32 region: q|k|v pre-activations, later reused as o | o_spike.
    size_t Foff = take((size_t)(2 * TBCN + TBCvN) * 4);
    float*    q_buf = (float*)(ws + Foff);
    float*    k_buf = (float*)(ws + Foff + (size_t)TBCN * 4);
    float*    v_buf = (float*)(ws + Foff + (size_t)2 * TBCN * 4);
    float*    o_buf = (float*)(ws + Foff);                        // alias
    _Float16* osT   = (_Float16*)(ws + Foff + (size_t)TBCvN * 4); // alias tail

    // --------------------- 1. weights f32 -> f16 --------------------------
    {
        long n = (long)C_DIM * C_DIM;
        cvt_f16_kernel<<<dim3((n + 255) / 256), 256, 0, stream>>>(q_w, wq16, n);
        cvt_f16_kernel<<<dim3((n + 255) / 256), 256, 0, stream>>>(k_w, wk16, n);
        n = (long)CV_DIM * C_DIM;
        cvt_f16_kernel<<<dim3((n + 255) / 256), 256, 0, stream>>>(v_w, wv16, n);
        n = (long)C_DIM * CV_DIM;
        cvt_f16_kernel<<<dim3((n + 255) / 256), 256, 0, stream>>>(p_w, wp16, n);
    }

    // --------------------- 2. LIF on x (+ pos) -> pixel-major spikes ------
    {
        long per_t = (long)B_DIM * CN;
        lif_kernel<true><<<dim3((per_t + 255) / 256), 256, 0, stream>>>(
            x_in, pos, xsT, per_t, C_DIM);
    }

    // --------------------- 3. q/k/v 1x1 conv + BN -------------------------
    gemm_wmma_kernel<false, true>
        <<<dim3(C_DIM / 128, N_PIX / 64, TB), 256, 0, stream>>>(
            wq16, 0, xsT, CN, 0, 1, q_buf, CN,
            C_DIM, C_DIM, C_DIM, N_PIX, q_s, q_b, 1.0f);
    gemm_wmma_kernel<false, true>
        <<<dim3(C_DIM / 128, N_PIX / 64, TB), 256, 0, stream>>>(
            wk16, 0, xsT, CN, 0, 1, k_buf, CN,
            C_DIM, C_DIM, C_DIM, N_PIX, k_s, k_b, 1.0f);
    gemm_wmma_kernel<false, true>
        <<<dim3(CV_DIM / 128, N_PIX / 64, TB), 256, 0, stream>>>(
            wv16, 0, xsT, CN, 0, 1, v_buf, CvN,
            C_DIM, C_DIM, C_DIM, N_PIX, v_s, v_b, 1.0f);

    // --------------------- 4. LIF on q/k/v --------------------------------
    {
        long per_t = (long)B_DIM * CN;
        lif_kernel<true><<<dim3((per_t + 255) / 256), 256, 0, stream>>>(
            q_buf, nullptr, qsT, per_t, C_DIM);
        lif_kernel<false><<<dim3((per_t + 255) / 256), 256, 0, stream>>>(
            k_buf, nullptr, ks, per_t, C_DIM);
        long per_tv = (long)B_DIM * CvN;
        lif_kernel<false><<<dim3((per_tv + 255) / 256), 256, 0, stream>>>(
            v_buf, nullptr, vs, per_tv, CV_DIM);
    }

    // --------------------- 5. vk[e,d] = sum_n v[e,n]*k[d,n] ---------------
    // z = ((t*B+b)*HEADS + h); A = v rows (lda=N); B = k rows as N-major.
    gemm_wmma_kernel<true, false>
        <<<dim3(DHV / 128, DH / 64, TBH), 256, 0, stream>>>(
            vs, (long)DHV * N_PIX,
            ks, (long)C_DIM * N_PIX, (long)DH * N_PIX, HEADS,
            vkb, (long)DHV * DH,
            N_PIX, N_PIX, N_PIX, DH, nullptr, nullptr, 1.0f);

    // --------------------- 6. o[e,n] = 0.25 * sum_d vk[e,d]*q[d,n] --------
    // B = q pixel-major: Bg[n*C + h*64 + d]  (split stride per tb / head).
    gemm_wmma_kernel<false, false>
        <<<dim3(DHV / 128, N_PIX / 64, TBH), 256, 0, stream>>>(
            vkb, (long)DHV * DH,
            qsT, (long)N_PIX * C_DIM, (long)DH, HEADS,
            o_buf, (long)DHV * N_PIX,
            DH, DH, C_DIM, N_PIX, nullptr, nullptr, 0.25f);

    // --------------------- 7. LIF on o -> pixel-major spikes --------------
    {
        long per_tv = (long)B_DIM * CvN;
        lif_kernel<true><<<dim3((per_tv + 255) / 256), 256, 0, stream>>>(
            o_buf, nullptr, osT, per_tv, CV_DIM);
    }

    // --------------------- 8. projection conv + BN -> out -----------------
    gemm_wmma_kernel<false, true>
        <<<dim3(C_DIM / 128, N_PIX / 64, TB), 256, 0, stream>>>(
            wp16, 0, osT, CvN, 0, 1, out, CN,
            CV_DIM, CV_DIM, CV_DIM, N_PIX, p_s, p_b, 1.0f);
}
